// MemEncoder_28570122453897
// MI455X (gfx1250) — matrix operand order repaired and checked by reading, not compiled, not run on hardware
//
#include <hip/hip_runtime.h>
#include <hip/hip_bf16.h>
#include <stdint.h>

typedef __attribute__((ext_vector_type(16))) _Float16 v16h;
typedef __attribute__((ext_vector_type(8)))  float    v8f;

// ---------------------------------------------------------------- utilities
__device__ inline float wave_sum(float x) {
#pragma unroll
  for (int o = 16; o; o >>= 1) x += __shfl_xor(x, o, 32);
  return x;
}
__device__ inline float wave_max(float x) {
#pragma unroll
  for (int o = 16; o; o >>= 1) x = fmaxf(x, __shfl_xor(x, o, 32));
  return x;
}

// ---------------------------------------------------------------- WMMA GEMM
// C[m,n] = alpha * sum_k A[m,k]*B[k,n]  (+ bias[n])
// Strided + 2-level batched: one kernel serves projections, per-(b,h)
// attention GEMMs and per-chunk bilinear Dense layers.
struct GemmP {
  const float* __restrict__ A;
  const float* __restrict__ Bw;
  const float* __restrict__ bias;
  float* __restrict__ C;
  int M, N, Kd;
  long long sAm, sAk, sBk, sBn, sCm;
  long long bAo, bAi, bBo, bBi, bCo, bCi, bBias;
  int innerB;
  float alpha;
};

// -------- pipelined staging helpers (global -> regs, regs -> LDS) --------
// LDS tiles are in contiguous-k WMMA operand order (k = 16*(lane/16) + elem, not the ISA table order; wave32):
//   A frag (16x32 f16): lane=(m&15)|((k>>4)<<4),   elem=k&15
//   B frag (32x16 f16): lane=(n&15)|((k>>4)<<4),   elem=k&15
// so each wave reads operands as contiguous 32B -> 2x ds_load_b128.

__device__ inline void loadA_regs(const float* __restrict__ A, float* rv, int tid,
    int row0, int k0, int M, int Kd, long long sAm, long long sAk, bool fast) {
#pragma unroll
  for (int i = 0; i < 4; ++i) {
    int idx = tid + i * 256;          // 0..1023 k-pairs
    int r = idx >> 4;                 // m_local 0..63
    int kl = (idx & 15) * 2;          // even k_local
    int gm = row0 + r, gk = k0 + kl;
    if (fast) {
      if (sAk == 1) {                 // every A in this model: contiguous k
        float2 t = *(const float2*)&A[(long long)gm * sAm + gk];
        rv[2 * i] = t.x; rv[2 * i + 1] = t.y;
      } else {
        long long base = (long long)gm * sAm;
        rv[2 * i]     = A[base + (long long)gk * sAk];
        rv[2 * i + 1] = A[base + (long long)(gk + 1) * sAk];
      }
    } else {                          // clamped, always-legal loads
      int gmc = gm < M ? gm : M - 1;
      int g0 = gk < Kd ? gk : Kd - 1;
      int g1 = gk + 1 < Kd ? gk + 1 : Kd - 1;
      long long base = (long long)gmc * sAm;
      rv[2 * i]     = A[base + (long long)g0 * sAk];
      rv[2 * i + 1] = A[base + (long long)g1 * sAk];
    }
  }
}

__device__ inline void storeA_lds(_Float16 (*sA)[32][16], const float* rv, int tid,
    int row0, int k0, int M, int Kd, bool fast) {
#pragma unroll
  for (int i = 0; i < 4; ++i) {
    int idx = tid + i * 256;
    int r = idx >> 4;
    int kl = (idx & 15) * 2;
    float v0 = rv[2 * i], v1 = rv[2 * i + 1];
    if (!fast) {                      // arithmetic mask keeps code branch-free
      int gm = row0 + r, gk = k0 + kl;
      v0 *= (gm < M && gk < Kd) ? 1.f : 0.f;
      v1 *= (gm < M && gk + 1 < Kd) ? 1.f : 0.f;
    }
    int la = (r & 15) | (((kl >> 4) & 1) << 4);
    int e  = kl & 15;                              // even -> pair in one dword
    union { _Float16 h[2]; uint32_t u; } pk;
    pk.h[0] = (_Float16)v0;
    pk.h[1] = (_Float16)v1;
    *(uint32_t*)&sA[r >> 4][la][e] = pk.u;
  }
}

__device__ inline void loadB_regs(const float* __restrict__ Bw, float* rv, int tid,
    int col0, int k0, int N, int Kd, long long sBk, long long sBn, bool fast) {
#pragma unroll
  for (int i = 0; i < 4; ++i) {
    int idx = tid + i * 256;          // 0..1023 n-pairs
    int r = idx >> 5;                 // k_local 0..31
    int c = (idx & 31) * 2;           // even n_local
    if (fast) {
      if (sBn == 1) {                 // weight matrices: contiguous n
        float2 t = *(const float2*)&Bw[(long long)(k0 + r) * sBk + (col0 + c)];
        rv[2 * i] = t.x; rv[2 * i + 1] = t.y;
      } else {
        long long base = (long long)(k0 + r) * sBk;
        rv[2 * i]     = Bw[base + (long long)(col0 + c) * sBn];
        rv[2 * i + 1] = Bw[base + (long long)(col0 + c + 1) * sBn];
      }
    } else {
      int gk = k0 + r;
      int gkc = gk < Kd ? gk : Kd - 1;
      int g0 = col0 + c, g1 = col0 + c + 1;
      int g0c = g0 < N ? g0 : N - 1;
      int g1c = g1 < N ? g1 : N - 1;
      long long base = (long long)gkc * sBk;
      rv[2 * i]     = Bw[base + (long long)g0c * sBn];
      rv[2 * i + 1] = Bw[base + (long long)g1c * sBn];
    }
  }
}

__device__ inline void storeB_lds(_Float16 (*sB)[32][16], const float* rv, int tid,
    int col0, int k0, int N, int Kd, bool fast) {
#pragma unroll
  for (int i = 0; i < 4; ++i) {
    int idx = tid + i * 256;
    int r = idx >> 5;
    int c = (idx & 31) * 2;           // even: c and c+1 share the 16-group
    float v0 = rv[2 * i], v1 = rv[2 * i + 1];
    if (!fast) {
      int gk = k0 + r;
      v0 *= (gk < Kd && (col0 + c) < N) ? 1.f : 0.f;
      v1 *= (gk < Kd && (col0 + c + 1) < N) ? 1.f : 0.f;
    }
    int la = (c & 15) | (((r >> 4) & 1) << 4);
    int e  = r & 15;
    sB[c >> 4][la][e]     = (_Float16)v0;
    sB[c >> 4][la + 1][e] = (_Float16)v1;
  }
}

__global__ __launch_bounds__(256) void gemm_wmma_f16(GemmP p) {
  __shared__ __align__(32) _Float16 sA[2][4][32][16];   // double buffered, 8KB
  __shared__ __align__(32) _Float16 sB[2][4][32][16];   // double buffered, 8KB
  const int tid  = threadIdx.x;
  const int lane = tid & 31;
  const int wid  = tid >> 5;        // 8 waves
  const int wm   = wid >> 1;        // 0..3 -> 16-row slice
  const int wn   = wid & 1;         // 0..1 -> 32-col slice
  const int row0 = blockIdx.y * 64;
  const int col0 = blockIdx.x * 64;
  const int zo = blockIdx.z / p.innerB;
  const int zi = blockIdx.z % p.innerB;
  const float* __restrict__ A  = p.A  + (long long)zo * p.bAo + (long long)zi * p.bAi;
  const float* __restrict__ Bw = p.Bw + (long long)zo * p.bBo + (long long)zi * p.bBi;
  float* __restrict__ C = p.C + (long long)zo * p.bCo + (long long)zi * p.bCi;
  const float* __restrict__ bias = p.bias ? (p.bias + (long long)zi * p.bBias) : nullptr;

  v8f acc0 = {};
  v8f acc1 = {};
  const int mloc = lane & 15;
  const int hi   = (lane >> 4) & 1;

  const bool aIntM = (row0 + 64 <= p.M);     // uniform
  const bool bIntN = (col0 + 64 <= p.N);     // uniform
  const int nk = (p.Kd + 31) >> 5;

  float ra[8], rb[8];
  // prologue: stage tile 0
  {
    bool kin = 32 <= p.Kd;
    loadA_regs(A, ra, tid, row0, 0, p.M, p.Kd, p.sAm, p.sAk, aIntM && kin);
    loadB_regs(Bw, rb, tid, col0, 0, p.N, p.Kd, p.sBk, p.sBn, bIntN && kin);
    storeA_lds(sA[0], ra, tid, row0, 0, p.M, p.Kd, aIntM && kin);
    storeB_lds(sB[0], rb, tid, col0, 0, p.N, p.Kd, bIntN && kin);
  }
  __syncthreads();

  for (int t = 0; t < nk; ++t) {
    const int cur = t & 1;
    const int k1 = (t + 1) << 5;
    const bool more = (t + 1 < nk);
    const bool aF = aIntM && (k1 + 32 <= p.Kd);
    const bool bF = bIntN && (k1 + 32 <= p.Kd);
    // issue next tile's global loads before this tile's math
    if (more) {
      loadA_regs(A, ra, tid, row0, k1, p.M, p.Kd, p.sAm, p.sAk, aF);
      loadB_regs(Bw, rb, tid, col0, k1, p.N, p.Kd, p.sBk, p.sBn, bF);
    }
    v16h a  = *(const v16h*)&sA[cur][wm][lane][0];
    v16h b0 = *(const v16h*)&sB[cur][wn * 2][lane][0];
    v16h b1 = *(const v16h*)&sB[cur][wn * 2 + 1][lane][0];
    acc0 = __builtin_amdgcn_wmma_f32_16x16x32_f16(false, a, false, b0, (short)0, acc0, false, false);
    acc1 = __builtin_amdgcn_wmma_f32_16x16x32_f16(false, a, false, b1, (short)0, acc1, false, false);
    // publish next tile after the WMMAs are issued
    if (more) {
      storeA_lds(sA[cur ^ 1], ra, tid, row0, k1, p.M, p.Kd, aF);
      storeB_lds(sB[cur ^ 1], rb, tid, col0, k1, p.N, p.Kd, bF);
    }
    __syncthreads();
  }

  // C/D layout: VGPR r -> M = r + 8*(lane>=16); N = lane&15
  const int mb  = wm * 16 + (hi ? 8 : 0);
  const int gn0 = col0 + wn * 32 + mloc;
  const int gn1 = gn0 + 16;
#pragma unroll
  for (int r = 0; r < 8; ++r) {
    int gm = row0 + mb + r;
    if (gm >= p.M) continue;
    if (gn0 < p.N) {
      float o = acc0[r] * p.alpha;
      if (bias) o += bias[gn0];
      C[(long long)gm * p.sCm + gn0] = o;
    }
    if (gn1 < p.N) {
      float o = acc1[r] * p.alpha;
      if (bias) o += bias[gn1];
      C[(long long)gm * p.sCm + gn1] = o;
    }
  }
}

// ---------------------------------------------------------------- gating
// One wave per (b,h,s): gk = k_row @ Wgk + bgk ; gq = q_row @ Wgq + bgq ;
// M = sigmoid((gk*gq) @ Wgm + bgm); q *= M0 ; k *= M1  (in place).
__global__ __launch_bounds__(256) void gate_kernel(
    float* q, float* k,
    const float* __restrict__ wgk, const float* __restrict__ bgk,
    const float* __restrict__ wgq, const float* __restrict__ bgq,
    const float* __restrict__ wgm, const float* __restrict__ bgm) {
  int w = blockIdx.x * 8 + (threadIdx.x >> 5);   // 0 .. 128*512-1
  int lane = threadIdx.x & 31;
  int s  = w & 511;
  int bh = w >> 9;
  int h  = bh & 7;
  int b  = bh >> 3;
  const long long base = ((long long)(b * 512 + s)) * 512 + h * 64;
  float* qr = q + base;
  float* kr = k + base;
  float acc0 = 0.f, acc1 = 0.f;
  for (int ii = 0; ii < 16; ++ii) {
    int i = lane + ii * 32;                       // gate index 0..511
    float gk = bgk[i], gq = bgq[i];
    for (int d = 0; d < 64; ++d) {
      gk += kr[d] * wgk[d * 512 + i];
      gq += qr[d] * wgq[d * 512 + i];
    }
    float gg = gk * gq;
    acc0 += gg * wgm[i * 2 + 0];
    acc1 += gg * wgm[i * 2 + 1];
  }
  acc0 = wave_sum(acc0);
  acc1 = wave_sum(acc1);
  float m0 = 1.f / (1.f + __expf(-(acc0 + bgm[0])));
  float m1 = 1.f / (1.f + __expf(-(acc1 + bgm[1])));
  for (int d = lane; d < 64; d += 32) { qr[d] *= m0; kr[d] *= m1; }
}

// ---------------------------------------------------------------- softmax
__global__ __launch_bounds__(256) void softmax_rows(
    float* x, const float* __restrict__ mask, int rows, int L, int S) {
  int w = blockIdx.x * 8 + (threadIdx.x >> 5);
  if (w >= rows) return;
  int lane = threadIdx.x & 31;
  float* row = x + (long long)w * L;
  const float* mrow = mask ? mask + (long long)(w % S) * L : nullptr;
  float mx = -3.0e38f;
  for (int i = lane; i < L; i += 32) {
    float v = row[i];
    if (mrow) v += mrow[i] * -1e9f;
    mx = fmaxf(mx, v);
  }
  mx = wave_max(mx);
  float sum = 0.f;
  for (int i = lane; i < L; i += 32) {
    float v = row[i];
    if (mrow) v += mrow[i] * -1e9f;
    float e = __expf(v - mx);
    row[i] = e;
    sum += e;
  }
  sum = wave_sum(sum);
  float inv = 1.f / sum;
  for (int i = lane; i < L; i += 32) row[i] *= inv;
}

// ---------------------------------------------------------------- add + LN
__global__ __launch_bounds__(256) void add_ln(
    const float* __restrict__ x, const float* __restrict__ res,
    const float* __restrict__ g, const float* __restrict__ bb,
    float* __restrict__ out, int rows) {
  int w = blockIdx.x * 8 + (threadIdx.x >> 5);
  if (w >= rows) return;
  int lane = threadIdx.x & 31;
  const float* xr = x + (long long)w * 512;
  const float* rr = res + (long long)w * 512;
  float vals[16];
  float s = 0.f;
#pragma unroll
  for (int i = 0; i < 16; ++i) {
    float v = xr[lane + 32 * i] + rr[lane + 32 * i];
    vals[i] = v; s += v;
  }
  s = wave_sum(s);
  float mean = s * (1.f / 512.f);
  float vs = 0.f;
#pragma unroll
  for (int i = 0; i < 16; ++i) { float d = vals[i] - mean; vs += d * d; }
  vs = wave_sum(vs);
  float inv = rsqrtf(vs * (1.f / 512.f) + 1e-6f);
#pragma unroll
  for (int i = 0; i < 16; ++i) {
    int c = lane + 32 * i;
    out[(long long)w * 512 + c] = (vals[i] - mean) * inv * g[c] + bb[c];
  }
}

// ------------------------------------------------- bilinear slice reduce
// z[(m*20+c)*15 + rk] = sum_{n<80} m0[(m*20+c)*80+n] * m1[...]
__global__ __launch_bounds__(256) void mul_reduce80(
    const float* __restrict__ m0, const float* __restrict__ m1,
    float* __restrict__ z, int rk) {
  int w = blockIdx.x * 8 + (threadIdx.x >> 5);   // token*20+chunk
  int lane = threadIdx.x & 31;
  const float* a = m0 + (long long)w * 80;
  const float* b = m1 + (long long)w * 80;
  float s = 0.f;
  for (int i = lane; i < 80; i += 32) s += a[i] * b[i];
  s = wave_sum(s);
  if (lane == 0) z[(long long)w * 15 + rk] = s;
}

// signed sqrt + per-chunk l2 normalize over RK=15
__global__ __launch_bounds__(256) void ssqrt_norm(float* z, int total) {
  int i = blockIdx.x * 256 + threadIdx.x;
  if (i >= total) return;
  float* zp = z + (long long)i * 15;
  float v[15]; float ss = 0.f;
#pragma unroll
  for (int r = 0; r < 15; ++r) {
    float x = zp[r];
    float s = (x >= 0.f) ? sqrtf(x) : -sqrtf(-x);
    v[r] = s; ss += s * s;
  }
  float inv = rsqrtf(fmaxf(ss, 1e-12f));
#pragma unroll
  for (int r = 0; r < 15; ++r) zp[r] = v[r] * inv;
}

// ---------------------------------------------------------------- elementwise
__global__ void ew_add(const float* __restrict__ a, const float* __restrict__ b,
                       float* __restrict__ o, long long n) {
  long long i = (long long)blockIdx.x * blockDim.x + threadIdx.x;
  if (i < n) o[i] = a[i] + b[i];
}
__global__ void ew_selu(float* x, long long n) {
  long long i = (long long)blockIdx.x * blockDim.x + threadIdx.x;
  if (i < n) {
    const float sc = 1.0507009873554805f, al = 1.6732632423543772f;
    float v = x[i];
    x[i] = v > 0.f ? sc * v : sc * al * (__expf(v) - 1.f);
  }
}

// ---------------------------------------------------------------- host side
static inline void launch_gemm(hipStream_t st,
    const float* A, const float* Bw, const float* bias, float* C,
    int M, int N, int Kd,
    long long sAm, long long sAk, long long sBk, long long sBn, long long sCm,
    long long bAo, long long bAi, long long bBo, long long bBi,
    long long bCo, long long bCi, long long bBias,
    int nOuter, int nInner, float alpha) {
  GemmP p{A, Bw, bias, C, M, N, Kd, sAm, sAk, sBk, sBn, sCm,
          bAo, bAi, bBo, bBi, bCo, bCi, bBias, nInner, alpha};
  dim3 grid((unsigned)((N + 63) / 64), (unsigned)((M + 63) / 64),
            (unsigned)(nOuter * nInner));
  gemm_wmma_f16<<<grid, 256, 0, st>>>(p);
}

static void run_mha(hipStream_t st,
    const float* src_q, const float* src_kv,
    const float* wq, const float* bq, const float* wk, const float* bk,
    const float* wv, const float* bv, const float* wo, const float* bo,
    const float* mask, bool gated,
    const float* gkw, const float* gkb, const float* gqw, const float* gqb,
    const float* gmw, const float* gmb,
    float* q, float* k, float* v, float* probs, float* tmp, float* out) {
  const int Bn = 16, S = 512, Dm = 512, Hh = 8;
  const int T = 8192;
  launch_gemm(st, src_q,  wq, bq, q, T, Dm, Dm, Dm,1, Dm,1, Dm, 0,0,0,0,0,0,0, 1,1, 1.f);
  launch_gemm(st, src_kv, wk, bk, k, T, Dm, Dm, Dm,1, Dm,1, Dm, 0,0,0,0,0,0,0, 1,1, 1.f);
  launch_gemm(st, src_kv, wv, bv, v, T, Dm, Dm, Dm,1, Dm,1, Dm, 0,0,0,0,0,0,0, 1,1, 1.f);
  if (gated)
    gate_kernel<<<8192, 256, 0, st>>>(q, k, gkw, gkb, gqw, gqb, gmw, gmb);
  // logits[b,h,qs,t] = 1/8 * q . k   (per-head batched, k used transposed)
  launch_gemm(st, q, k, nullptr, probs, S, S, 64,
      Dm, 1, 1, Dm, S,
      (long long)S * Dm, 64, (long long)S * Dm, 64,
      (long long)Hh * S * S, (long long)S * S, 0, Bn, Hh, 0.125f);
  softmax_rows<<<8192, 256, 0, st>>>(probs, mask, Bn * Hh * S, S, S);
  // out[b,s,h*64+d] = probs . v
  launch_gemm(st, probs, v, nullptr, tmp, S, 64, S,
      S, 1, Dm, 1, Dm,
      (long long)Hh * S * S, (long long)S * S, (long long)S * Dm, 64,
      (long long)S * Dm, 64, 0, Bn, Hh, 1.f);
  launch_gemm(st, tmp, wo, bo, out, T, Dm, Dm, Dm,1, Dm,1, Dm, 0,0,0,0,0,0,0, 1,1, 1.f);
}

static void run_block(hipStream_t st, const float* xa, const float* xb,
    const float* l0w, const float* l0b, const float* l1w, const float* l1b,
    const float* m0w, const float* m0b, const float* m1w, const float* m1b,
    const float* outw, const float* outb,
    float* x0, float* x1, float* m0buf, float* m1buf, float* z, float* out) {
  const int T = 8192;
  launch_gemm(st, xa, l0w, l0b, x0, T, 1600, 512, 512,1, 1600,1, 1600, 0,0,0,0,0,0,0, 1,1, 1.f);
  launch_gemm(st, xb, l1w, l1b, x1, T, 1600, 512, 512,1, 1600,1, 1600, 0,0,0,0,0,0,0, 1,1, 1.f);
  for (int rk = 0; rk < 15; ++rk) {
    // per-chunk Dense(80 -> 80) slice rk, batched over 20 chunks
    launch_gemm(st, x0, m0w + (long long)rk * 80, m0b + (long long)rk * 80, m0buf,
        T, 80, 80,
        1600, 1, 1200, 1, 1600,
        0, 80, 0, 96000, 0, 80, 1200,
        1, 20, 1.f);
    launch_gemm(st, x1, m1w + (long long)rk * 80, m1b + (long long)rk * 80, m1buf,
        T, 80, 80,
        1600, 1, 1200, 1, 1600,
        0, 80, 0, 96000, 0, 80, 1200,
        1, 20, 1.f);
    mul_reduce80<<<20480, 256, 0, st>>>(m0buf, m1buf, z, rk);
  }
  ssqrt_norm<<<(T * 20 + 255) / 256, 256, 0, st>>>(z, T * 20);
  launch_gemm(st, z, outw, outb, out, T, 512, 300, 300,1, 512,1, 512, 0,0,0,0,0,0,0, 1,1, 1.f);
}

#define PARAM(i) ((const float*)d_in[6 + (i)])

extern "C" void kernel_launch(void* const* d_in, const int* in_sizes, int n_in,
                              void* d_out, int out_size, void* d_ws, size_t ws_size,
                              hipStream_t stream) {
  (void)in_sizes; (void)n_in; (void)out_size; (void)ws_size;
  const long long T = 8192;

  const float* mem   = (const float*)d_in[0];
  const float* ctrl  = (const float*)d_in[1];
  const float* words = (const float*)d_in[2];
  const float* kbase = (const float*)d_in[3];
  const float* mask  = (const float*)d_in[4];
  // d_in[5] padding_mask: unused (reference passes mask=None to mha2)

  float* Wp = (float*)d_ws;
  long long off = 0;
  auto take = [&](long long n) { float* r = Wp + off; off += n; return r; };
  float* q     = take(T * 512);
  float* kbuf  = take(T * 512);
  float* vbuf  = take(T * 512);
  float* probs = take((long long)128 * 512 * 512);
  float* t0    = take(T * 512);
  float* out1  = take(T * 512);
  float* out2  = take(T * 512);
  float* out3  = take(T * 512);
  float* comb  = take(T * 512);
  float* x0    = take(T * 1600);
  float* x1    = take(T * 1600);
  float* m0b_  = take(T * 1600);
  float* m1b_  = take(T * 1600);
  float* zb    = take(T * 300);
  float* ffnb  = take(T * 2048);

  // ---- gated self attention over mem_state + LN1
  run_mha(stream, mem, mem,
          PARAM(0), PARAM(1), PARAM(2), PARAM(3), PARAM(4), PARAM(5), PARAM(6), PARAM(7),
          mask, true,
          PARAM(16), PARAM(17), PARAM(18), PARAM(19), PARAM(20), PARAM(21),
          q, kbuf, vbuf, probs, t0, comb);
  add_ln<<<1024, 256, 0, stream>>>(comb, mem, PARAM(22), PARAM(23), out1, 8192);

  // ---- cross attention into knowledge base + LN2
  run_mha(stream, out1, kbase,
          PARAM(8), PARAM(9), PARAM(10), PARAM(11), PARAM(12), PARAM(13), PARAM(14), PARAM(15),
          nullptr, false,
          nullptr, nullptr, nullptr, nullptr, nullptr, nullptr,
          q, kbuf, vbuf, probs, t0, comb);
  add_ln<<<1024, 256, 0, stream>>>(comb, out1, PARAM(24), PARAM(25), out2, 8192);

  // ---- bilinear block 1 (out2, control_state) + LN3
  run_block(stream, out2, ctrl,
            PARAM(34), PARAM(35), PARAM(36), PARAM(37),
            PARAM(38), PARAM(39), PARAM(40), PARAM(41),
            PARAM(42), PARAM(43),
            x0, x1, m0b_, m1b_, zb, t0);
  add_ln<<<1024, 256, 0, stream>>>(t0, out2, PARAM(26), PARAM(27), out3, 8192);

  // ---- bilinear block 2 (out3, attend_words); comb = z2 + out3
  run_block(stream, out3, words,
            PARAM(44), PARAM(45), PARAM(46), PARAM(47),
            PARAM(48), PARAM(49), PARAM(50), PARAM(51),
            PARAM(52), PARAM(53),
            x0, x1, m0b_, m1b_, zb, t0);
  ew_add<<<16384, 256, 0, stream>>>(t0, out3, comb, T * 512);

  // ---- FFN (selu) + LN4 -> output
  launch_gemm(stream, comb, PARAM(30), PARAM(31), ffnb,
              8192, 2048, 512, 512,1, 2048,1, 2048, 0,0,0,0,0,0,0, 1,1, 1.f);
  ew_selu<<<65536, 256, 0, stream>>>(ffnb, T * 2048);
  launch_gemm(stream, ffnb, PARAM(32), PARAM(33), t0,
              8192, 512, 2048, 2048,1, 512,1, 512, 0,0,0,0,0,0,0, 1,1, 1.f);
  add_ln<<<1024, 256, 0, stream>>>(t0, comb, PARAM(28), PARAM(29), (float*)d_out, 8192);
}
